// mapaa_52286931861674
// MI455X (gfx1250) — matrix operand order repaired and checked by reading, not compiled, not run on hardware
//
#include <hip/hip_runtime.h>
#include <hip/hip_bf16.h>

// ---------------------------------------------------------------------------
// Fused 1x1-conv GEMM (bf16 WMMA) + cosine-similarity loss for MI455X gfx1250.
//   M = N*H*W = 100352, K = 2048, N = 256; feat never hits HBM.
//   B (weight) panels: double-buffered LDS via global_load_async_to_lds_b128.
//   All 8 B fragments kept live so the 8 WMMAs issue back-to-back.
// ---------------------------------------------------------------------------

typedef __attribute__((ext_vector_type(8)))  __bf16 bf16x8;
typedef __attribute__((ext_vector_type(16))) __bf16 bf16x16;
typedef __attribute__((ext_vector_type(8)))  float  f32x8;

#define AS3 __attribute__((address_space(3)))

#define N_SAMP 2048
#define C_IN   2048
#define C_OUT  256
#define HW     49
#define LROWS  1000
#define KSTEP  32
#define NKSTEP (C_IN / KSTEP)     // 64
#define A_STRIDE 40               // bf16 elems per A row (32 + pad); 80 B, 16B-multiple
#define PANEL_BYTES (C_OUT * KSTEP * 2)   // 16384 B per K-step weight panel

struct alignas(16) SMem {
  __bf16 A[64 * A_STRIDE];        // 5120 B : 64 (padded M) x 32 (K) bf16, row-major
  __bf16 B[2][C_OUT * KSTEP];     // 2 x 16384 B: 256 (o) x 32 (K) bf16, o-major
  float  rows[C_OUT];
  float  bias[C_OUT];
  float  dsum[64];
  float  ssum[64];
  float  sumsq;
  float  per;
  int    nonzero;
  float  nr;
  float  maskf;
};

__device__ __forceinline__ bf16x16 load_frag(const __bf16* p) {
  bf16x8 lo = *(const bf16x8*)(p);
  bf16x8 hi = *(const bf16x8*)(p + 8);
  return __builtin_shufflevector(lo, hi, 0,1,2,3,4,5,6,7,8,9,10,11,12,13,14,15);
}

// Pre-pack W_conv (256 x 2048 f32, o-major) into bf16 K-step panels:
//   wpack[kc][o][klo] -> each K-step panel is one contiguous 16 KB block.
__global__ __launch_bounds__(256) void prep_w_kernel(const float* __restrict__ W,
                                                     __bf16* __restrict__ wpack) {
  int e   = blockIdx.x * 256 + threadIdx.x;   // 2048 blocks * 256 = 524288 elements
  int kc  = e >> 13;
  int rem = e & 8191;
  int o   = rem >> 5;
  int klo = rem & 31;
  wpack[e] = (__bf16)W[o * C_IN + (kc * KSTEP + klo)];
}

__global__ __launch_bounds__(256) void fused_main(
    const float* __restrict__ feature, const float* __restrict__ input,
    const float* __restrict__ lut, const int* __restrict__ label,
    const float* __restrict__ b_conv, const __bf16* __restrict__ wpack,
    float* __restrict__ per_n)
{
  __shared__ SMem sm;
  const int tid  = threadIdx.x;
  const int n    = blockIdx.x;
  const int lane = tid & 31;
  const int wave = tid >> 5;
  const int mt   = wave & 3;     // M-tile of this wave
  const int ng   = wave >> 2;    // which half of the 16 N-tiles
  const int col  = lane & 15;    // N index within tile
  const int hi   = lane >> 4;    // lane half (K/M sub-block select per WMMA layout)

  const float* feat_n = feature + (size_t)n * C_IN * HW;

  // LDS byte offsets of the two B buffers (raw LDS addresses for async DMA vdst)
  unsigned ldsB[2];
  ldsB[0] = (unsigned)(unsigned long long)(AS3 char*)(&sm.B[0][0]);
  ldsB[1] = (unsigned)(unsigned long long)(AS3 char*)(&sm.B[1][0]);

  // ---- per-sample metadata: rows = lut[label-1], nr, mask, bias ----
  int lab  = label[n] - 1;
  int safe = lab < 0 ? 0 : (lab > LROWS - 1 ? LROWS - 1 : lab);
  float rv = lut[(size_t)safe * C_OUT + tid];
  sm.rows[tid] = rv;
  sm.bias[tid] = b_conv[tid];
  if (tid == 0) { sm.sumsq = 0.f; sm.per = 0.f; sm.nonzero = 0; }
  if (tid < 64) { sm.dsum[tid] = 0.f; sm.ssum[tid] = 0.f; }
  for (int e = tid; e < 64 * A_STRIDE; e += 256) sm.A[e] = (__bf16)0.0f; // zero pad rows
  __syncthreads();
  atomicAdd(&sm.sumsq, rv * rv);
  if (rv != 0.0f) atomicOr(&sm.nonzero, 1);
  __syncthreads();
  if (tid == 0) {
    sm.nr    = sqrtf(sm.sumsq);
    sm.maskf = (lab >= 0 && lab < LROWS && sm.nonzero) ? 1.0f : 0.0f;
  }

  // ---- loop-invariant A-staging offsets: pack 2 channels per ds_store_b32 ----
  // pair u in [0,784): cpair = u/49, p = u%49 ; global offs (2c*49+p), LDS off p*A_STRIDE+2c
  int goffA[4], aoffA[4];
#pragma unroll
  for (int i = 0; i < 4; ++i) {
    int u  = tid + i * 256;
    int cp = u / HW;
    int p  = u - cp * HW;
    goffA[i] = (2 * cp) * HW + p;
    aoffA[i] = p * A_STRIDE + 2 * cp;
  }

  // ---- async-DMA helper: one 16 KB weight panel, 4 x b128 per lane ----
  const unsigned long long wbase = (unsigned long long)(const char*)wpack;
  auto issueB = [&](int kc, int buf) {
    unsigned            l = ldsB[buf] + (unsigned)(tid * 16);
    unsigned long long  g = wbase + (unsigned long long)kc * PANEL_BYTES + (unsigned)(tid * 16);
#pragma unroll
    for (int i = 0; i < 4; ++i) {
      asm volatile("global_load_async_to_lds_b128 %0, %1, off"
                   :: "v"(l + (unsigned)(i * 4096)),
                      "v"(g + (unsigned long long)(i * 4096))
                   : "memory");
    }
  };

  f32x8 acc[8] = {};
  const __bf16* aptr = &sm.A[(mt * 16 + col) * A_STRIDE + hi * 16];

  issueB(0, 0);                 // prologue: panel 0 in flight
  __syncthreads();              // metadata visible before main loop

  for (int kc = 0; kc < NKSTEP; ++kc) {
    // Stage A: contiguous 6272 B slab feature[n][32c..][..] -> LDS [p][k] bf16 (packed pairs)
    const float* gsrc = feat_n + (size_t)kc * (KSTEP * HW);
#pragma unroll
    for (int i = 0; i < 4; ++i) {
      if (i < 3 || tid < 16) {            // 784 channel-pairs total
        float f0 = gsrc[goffA[i]];
        float f1 = gsrc[goffA[i] + HW];
        unsigned lo16 = (unsigned)__builtin_bit_cast(unsigned short, (__bf16)f0);
        unsigned hi16 = (unsigned)__builtin_bit_cast(unsigned short, (__bf16)f1);
        *(unsigned*)(&sm.A[aoffA[i]]) = lo16 | (hi16 << 16);
      }
    }
    if (kc + 1 < NKSTEP) {
      issueB(kc + 1, (kc + 1) & 1);       // overlap next panel DMA with this K-step
      if (tid < HW)                        // gfx1250 global_prefetch_b8 on next A slab
        __builtin_prefetch((const char*)(feat_n + (size_t)(kc + 1) * (KSTEP * HW)) + tid * 128, 0, 0);
      asm volatile("s_wait_asynccnt 0x4" ::: "memory");  // panel kc landed; kc+1 in flight
    } else {
      asm volatile("s_wait_asynccnt 0x0" ::: "memory");
    }
    __syncthreads();

    // Load ALL fragments first (one DS clause), then fire 8 WMMAs back-to-back.
    // A fragment: lane<16 -> K{0..15}; lane>=16 -> K{16..31}
    const __bf16* bbase = &sm.B[kc & 1][(ng * 8 * 16 + col) * KSTEP + hi * 16];
    bf16x16 bfr[8];
#pragma unroll
    for (int j = 0; j < 8; ++j) bfr[j] = load_frag(bbase + j * (16 * KSTEP));
    bf16x16 afrag = load_frag(aptr);

#pragma unroll
    for (int j = 0; j < 8; ++j) {
      acc[j] = __builtin_amdgcn_wmma_f32_16x16x32_bf16(
          false, afrag, false, bfr[j], (short)0, acc[j], false, false);
    }
    __syncthreads();
  }

  // ---- fused epilogue: per-position dot(feat, rows) and ||feat||^2 ----
  float dloc[8], sloc[8];
#pragma unroll
  for (int rI = 0; rI < 8; ++rI) { dloc[rI] = 0.f; sloc[rI] = 0.f; }
#pragma unroll
  for (int j = 0; j < 8; ++j) {
    int   o  = (ng * 8 + j) * 16 + col;
    float rw = sm.rows[o];
    float bo = sm.bias[o];
#pragma unroll
    for (int rI = 0; rI < 8; ++rI) {
      float f = acc[j][rI] + bo;          // D layout: p = mt*16 + hi*8 + rI, channel = o
      dloc[rI] += f * rw;
      sloc[rI] += f * f;
    }
  }
#pragma unroll
  for (int rI = 0; rI < 8; ++rI) {        // reduce over the 16 lanes sharing each p
#pragma unroll
    for (int m2 = 1; m2 < 16; m2 <<= 1) {
      dloc[rI] += __shfl_xor(dloc[rI], m2, 32);
      sloc[rI] += __shfl_xor(sloc[rI], m2, 32);
    }
  }
  if (col == 0) {
    int pbase = mt * 16 + hi * 8;
#pragma unroll
    for (int rI = 0; rI < 8; ++rI) {
      atomicAdd(&sm.dsum[pbase + rI], dloc[rI]);
      atomicAdd(&sm.ssum[pbase + rI], sloc[rI]);
    }
  }
  __syncthreads();
  if (tid < HW) {
    float nf    = sqrtf(sm.ssum[tid]);
    float denom = fmaxf(nf * sm.nr, 1e-8f);
    float cosv  = sm.dsum[tid] / denom;
    float z     = input[(size_t)n * HW + tid];
    atomicAdd(&sm.per, fabsf(cosv - z));
  }
  __syncthreads();
  if (tid == 0) per_n[n] = sm.maskf * sm.per * (1.0f / 324.0f);
}

__global__ __launch_bounds__(256) void reduce_kernel(const float* __restrict__ per_n,
                                                     float* __restrict__ out) {
  __shared__ float s[256];
  float v = 0.f;
  for (int i = threadIdx.x; i < N_SAMP; i += 256) v += per_n[i];
  s[threadIdx.x] = v;
  __syncthreads();
  for (int st = 128; st > 0; st >>= 1) {
    if ((int)threadIdx.x < st) s[threadIdx.x] += s[threadIdx.x + st];
    __syncthreads();
  }
  if (threadIdx.x == 0) out[0] = s[0] + 0.01f;
}

extern "C" void kernel_launch(void* const* d_in, const int* in_sizes, int n_in,
                              void* d_out, int out_size, void* d_ws, size_t ws_size,
                              hipStream_t stream) {
  const float* feature = (const float*)d_in[0];
  const float* input   = (const float*)d_in[1];
  const float* lut     = (const float*)d_in[2];
  const int*   label   = (const int*)d_in[3];
  const float* W_conv  = (const float*)d_in[4];
  const float* b_conv  = (const float*)d_in[5];

  __bf16* wpack = (__bf16*)d_ws;                                    // 1 MiB
  float*  per_n = (float*)((char*)d_ws + (size_t)C_OUT * C_IN * 2); // 8 KiB

  prep_w_kernel<<<2048, 256, 0, stream>>>(W_conv, wpack);
  fused_main<<<N_SAMP, 256, 0, stream>>>(feature, input, lut, label, b_conv, wpack, per_n);
  reduce_kernel<<<1, 256, 0, stream>>>(per_n, (float*)d_out);
}
